// BAESNN_46213848105974
// MI455X (gfx1250) — matrix operand order repaired and checked by reading, not compiled, not run on hardware
//
#include <hip/hip_runtime.h>

// ---------------------------------------------------------------------------
// BAESNN fused spiking forward + STDP outer-product accumulation for gfx1250.
// All GEMMs (forward levels and the dw einsums) run on v_wmma_f32_16x16x32_f16.
// Spikes/weights are exactly representable in f16; f32 accumulators are exact
// (integer-valued, < 2^24), so results are bit-identical to the reference and
// deterministic regardless of reduction order.
// ---------------------------------------------------------------------------

typedef __attribute__((ext_vector_type(16))) _Float16     v16h;
typedef __attribute__((ext_vector_type(8)))  float        v8f;
typedef __attribute__((ext_vector_type(4)))  float        f32x4;
typedef __attribute__((ext_vector_type(4)))  unsigned int u32x4;

union F16x16 { v16h h; u32x4 q[2]; _Float16 e[16]; };
union F16x8  { u32x4 q; _Float16 e[8]; };

#define WPB      2      // waves per block
#define TPB      64     // threads per block
#define NBLOCKS  1024
#define DWN      2880   // 48*50 + 48*10 stacked dw values per wave

__device__ __forceinline__ v16h ld2(const _Float16* p0, const _Float16* p1) {
  F16x16 u;
  u.q[0] = *(const u32x4*)p0;   // ds_load_b128
  u.q[1] = *(const u32x4*)p1;   // ds_load_b128
  return u.h;
}

__device__ __forceinline__ v8f wmma_f16(v16h a, v16h b, v8f c) {
  // D = A(16x32 f16) * B(32x16 f16) + C(16x16 f32)
  return __builtin_amdgcn_wmma_f32_16x16x32_f16(
      /*neg_a=*/false, a, /*neg_b=*/false, b,
      /*c_mod=*/(short)0, c, /*reuse_a=*/false, /*reuse_b=*/false);
}

__global__ __launch_bounds__(TPB, 1) void baesnn_main(
    const float* __restrict__ x1, const float* __restrict__ x2,
    const float* __restrict__ W0, const float* __restrict__ W1,
    const float* __restrict__ W2, const float* __restrict__ W3,
    const float* __restrict__ W4, const float* __restrict__ W5,
    const float* __restrict__ W6,
    float* __restrict__ out, float* __restrict__ partial,
    int useAtomic, int Bsz, int nWaves)
{
  // Block-shared pre-swizzled weight B-fragments (15 frags x 32 lanes x 16 f16)
  __shared__ __align__(16) _Float16 s_frag[15 * 512];
  // Per-wave staging (each wave uses only its own slice -> no intra-loop barriers;
  // DS ops are in-order within a wave, the compiler inserts the dscnt waits)
  __shared__ __align__(16) _Float16 s_s   [WPB][32][64]; // stacked spikes [m|p], row-major, K-padded
  __shared__ __align__(16) _Float16 s_sT  [WPB][48][32]; // stacked spikes transposed (i, b)
  __shared__ __align__(16) _Float16 s_ifgT[WPB][64][32]; // out_ifg transposed (o, b), o padded to 64
  __shared__ __align__(16) _Float16 s_smaT[WPB][16][32]; // out_sma transposed (o, b)
  __shared__ __align__(16) _Float16 s_sma [WPB][32][16]; // out_sma row-major

  const int lane  = threadIdx.x & 31;
  const int widx  = threadIdx.x >> 5;
  const int wglob = blockIdx.x * WPB + widx;
  const int l16   = lane & 15;
  const int lhalf = lane >> 4;
  const int kA    = lhalf * 8;   // A-fragment K run offset / D-row offset
  const int kB    = lhalf * 16;  // B-fragment K offset

  const u32x4 u4z = {0, 0, 0, 0};
  const f32x4 f4z = {0.f, 0.f, 0.f, 0.f};
  const v8f   cz  = {0.f, 0.f, 0.f, 0.f, 0.f, 0.f, 0.f, 0.f};

  // ---- build f16 weight B-fragments in LDS (ISA B layout: lane=N, K=8*(lane>=16)+e+8*(e>=8)) ----
  for (int t = threadIdx.x; t < 15 * 512; t += TPB) {
    const int f = t >> 9, r = t & 511;
    const int L = r >> 4, e = r & 15;
    const int k = ((L >> 4) << 3) + e + ((e >> 3) << 3);
    const int n = L & 15;
    float w = 0.0f;
    if (f < 2)      { int nn = f * 16 + n;       if (k < 24 && nn < 24) w = W0[k * 24 + nn]; }
    else if (f < 4) { int nn = (f - 2) * 16 + n; if (k < 24 && nn < 24) w = W3[k * 24 + nn]; }
    else if (f < 12) {
      const int kc = (f - 4) >> 2, nt = (f - 4) & 3;
      const int ii = kc * 32 + k, nn = nt * 16 + n;   // stacked [W1;W2], K=48 pad 64
      if (ii < 48 && nn < 50) w = (ii < 24) ? W1[ii * 50 + nn] : W2[(ii - 24) * 50 + nn];
    } else if (f < 14) {
      const int ii = (f - 12) * 32 + k;               // stacked [W4;W5]
      if (ii < 48 && n < 10) w = (ii < 24) ? W4[ii * 10 + n] : W5[(ii - 24) * 10 + n];
    } else {
      if (k < 10 && n < 10) w = W6[k * 10 + n];
    }
    s_frag[t] = (_Float16)w;
  }
  // zero K-padding columns 48..63 of this wave's row-major spike tile (written once;
  // cols 48..55 are re-written with zeros every chunk, 56..63 are never written)
  {
    _Float16* z = &s_s[widx][lane][48];
    ((u32x4*)z)[0] = u4z;
    ((u32x4*)z)[1] = u4z;
  }
  __syncthreads();

  auto WF = [&](int f) -> v16h {
    const _Float16* p = &s_frag[f * 512 + lane * 16];
    return ld2(p, p + 8);
  };

  _Float16 (*sS)[64]   = s_s[widx];
  _Float16 (*sT)[32]   = s_sT[widx];
  _Float16 (*ifgT)[32] = s_ifgT[widx];
  _Float16 (*smaT)[32] = s_smaT[widx];
  _Float16 (*sSma)[16] = s_sma[widx];

  float* __restrict__ oifg = out + DWN;
  float* __restrict__ osma = oifg + (size_t)Bsz * 50;
  float* __restrict__ om1  = osma + (size_t)Bsz * 10;

  // dw accumulators: stacked M=48 (i<24 -> *_m, i>=24 -> *_p), chained through C
  v8f accI[3][4];
  v8f accS[3];
#pragma unroll
  for (int mt = 0; mt < 3; ++mt) {
    accS[mt] = cz;
#pragma unroll
    for (int nt = 0; nt < 4; ++nt) accI[mt][nt] = cz;
  }

  const int nChunks = Bsz >> 5;  // 32 batch rows per chunk
  for (int c = wglob; c < nChunks; c += nWaves) {
    const int base = c << 5;

    // ---------------- level 0: out_m = spike(x1@W0), out_p = spike(x2@W3) ----------------
#pragma unroll
    for (int mp = 0; mp < 2; ++mp) {           // m first, then p (p overwrites m's N-pad cols)
      const float* xp  = mp ? x2 : x1;
      const int cshift = mp ? 24 : 0;
      const int fbase  = mp ? 2 : 0;
#pragma unroll
      for (int rt = 0; rt < 2; ++rt) {
        const float* row = xp + (size_t)(base + rt * 16 + l16) * 24;  // 96B rows -> 16B aligned
        F16x16 a;
        f32x4 p0 = ((const f32x4*)row)[lhalf * 2];
        f32x4 p1 = ((const f32x4*)row)[lhalf * 2 + 1];
        f32x4 p2 = f4z, p3 = f4z;
        if (!lhalf) { p2 = ((const f32x4*)row)[4]; p3 = ((const f32x4*)row)[5]; }
#pragma unroll
        for (int v = 0; v < 4; ++v) {
          a.e[v]      = (_Float16)p0[v];
          a.e[4 + v]  = (_Float16)p1[v];
          a.e[8 + v]  = (_Float16)p2[v];
          a.e[12 + v] = (_Float16)p3[v];
        }
        const int b0 = rt * 16 + kA;
#pragma unroll
        for (int nt = 0; nt < 2; ++nt) {
          v8f d = wmma_f16(a.h, WF(fbase + nt), cz);
          F16x8 sp;
#pragma unroll
          for (int v = 0; v < 8; ++v)
            sp.e[v] = d[v] > 0.5f ? (_Float16)1.0f : (_Float16)0.0f;
          const int icol = cshift + nt * 16 + l16;          // stacked i index
          if (icol < 48) *(u32x4*)&sT[icol][b0] = sp.q;     // transposed: 1x ds_store_b128
#pragma unroll
          for (int v = 0; v < 8; ++v) sS[b0 + v][icol] = sp.e[v];  // row-major
        }
      }
    }

    // ------- level 1: out_ifg = spike(s@[W1;W2]), out_sma = spike(s@[W4;W5]), K=48->2 chunks -------
#pragma unroll
    for (int rt = 0; rt < 2; ++rt) {
      const int rrow = rt * 16 + l16;
      v16h aK0 = ld2(&sS[rrow][kA],      &sS[rrow][16 + kA]);
      v16h aK1 = ld2(&sS[rrow][32 + kA], &sS[rrow][48 + kA]);
      const int b0 = rt * 16 + kA;
#pragma unroll
      for (int nt = 0; nt < 4; ++nt) {
        v8f d = wmma_f16(aK0, WF(4 + nt), cz);
        d     = wmma_f16(aK1, WF(8 + nt), d);
        F16x8 sp;
#pragma unroll
        for (int v = 0; v < 8; ++v)
          sp.e[v] = d[v] > 0.5f ? (_Float16)1.0f : (_Float16)0.0f;
        const int o = nt * 16 + l16;
        *(u32x4*)&ifgT[o][b0] = sp.q;
        if (o < 50) {
#pragma unroll
          for (int v = 0; v < 8; ++v)
            oifg[(size_t)(base + b0 + v) * 50 + o] = d[v] > 0.5f ? 1.0f : 0.0f;
        }
      }
      {
        v8f d = wmma_f16(aK0, WF(12), cz);
        d     = wmma_f16(aK1, WF(13), d);
        F16x8 sp;
#pragma unroll
        for (int v = 0; v < 8; ++v)
          sp.e[v] = d[v] > 0.5f ? (_Float16)1.0f : (_Float16)0.0f;
        *(u32x4*)&smaT[l16][b0] = sp.q;
#pragma unroll
        for (int v = 0; v < 8; ++v) sSma[b0 + v][l16] = sp.e[v];
        if (l16 < 10) {
#pragma unroll
          for (int v = 0; v < 8; ++v)
            osma[(size_t)(base + b0 + v) * 10 + l16] = d[v] > 0.5f ? 1.0f : 0.0f;
        }
      }
    }

    // ---------------- level 2: out_m1 = spike(out_sma @ W6) ----------------
#pragma unroll
    for (int rt = 0; rt < 2; ++rt) {
      F16x16 a;
      a.q[0] = *(const u32x4*)&sSma[rt * 16 + l16][kA];  // K 0..15 (cols>=10 are zero)
      a.q[1] = u4z;                                      // K 16..31 pad
      v8f d = wmma_f16(a.h, WF(14), cz);
      if (l16 < 10) {
        const int b0 = rt * 16 + kA;
#pragma unroll
        for (int v = 0; v < 8; ++v)
          om1[(size_t)(base + b0 + v) * 10 + l16] = d[v] > 0.5f ? 1.0f : 0.0f;
      }
    }

    // ------- dw einsums: acc[i,o] += sum_b s[b,i]*post[b,o]  (A = s^T, K = 32 batch rows) -------
    v16h bI0 = ld2(&ifgT[ 0 + l16][kB], &ifgT[ 0 + l16][kB + 8]);
    v16h bI1 = ld2(&ifgT[16 + l16][kB], &ifgT[16 + l16][kB + 8]);
    v16h bI2 = ld2(&ifgT[32 + l16][kB], &ifgT[32 + l16][kB + 8]);
    v16h bI3 = ld2(&ifgT[48 + l16][kB], &ifgT[48 + l16][kB + 8]);
    v16h bS  = ld2(&smaT[l16][kB],      &smaT[l16][kB + 8]);
#pragma unroll
    for (int mt = 0; mt < 3; ++mt) {
      v16h aT = ld2(&sT[mt * 16 + l16][kB], &sT[mt * 16 + l16][kB + 8]);
      accI[mt][0] = wmma_f16(aT, bI0, accI[mt][0]);
      accI[mt][1] = wmma_f16(aT, bI1, accI[mt][1]);
      accI[mt][2] = wmma_f16(aT, bI2, accI[mt][2]);
      accI[mt][3] = wmma_f16(aT, bS == bS ? bI3 : bI3, accI[mt][3]); // keep simple: bI3
      accS[mt]    = wmma_f16(aT, bS, accS[mt]);
    }
  }

  // ---------------- dw write-out ----------------
  if (!useAtomic) {
    float* pw = partial + (size_t)wglob * DWN;   // exclusive slice, plain stores
#pragma unroll
    for (int mt = 0; mt < 3; ++mt) {
#pragma unroll
      for (int v = 0; v < 8; ++v) {
        const int i = mt * 16 + kA + v;
#pragma unroll
        for (int nt = 0; nt < 4; ++nt) {
          const int o = nt * 16 + l16;
          if (o < 50) pw[i * 50 + o] = accI[mt][nt][v];
        }
        if (l16 < 10) pw[2400 + i * 10 + l16] = accS[mt][v];
      }
    }
  } else {
#pragma unroll
    for (int mt = 0; mt < 3; ++mt) {
#pragma unroll
      for (int v = 0; v < 8; ++v) {
        const int i = mt * 16 + kA + v;
#pragma unroll
        for (int nt = 0; nt < 4; ++nt) {
          const int o = nt * 16 + l16;
          if (o < 50) {
            float* dst = (i < 24) ? (out + i * 50 + o) : (out + 1200 + (i - 24) * 50 + o);
            atomicAdd(dst, accI[mt][nt][v]);   // integer-valued -> exact & deterministic
          }
        }
        if (l16 < 10) {
          float* dst = (i < 24) ? (out + 2400 + i * 10 + l16) : (out + 2640 + (i - 24) * 10 + l16);
          atomicAdd(dst, accS[mt][v]);
        }
      }
    }
  }
}

__global__ void baesnn_reduce(const float* __restrict__ part, float* __restrict__ out, int nWaves) {
  const int j = blockIdx.x * blockDim.x + threadIdx.x;
  if (j >= DWN) return;
  float s = 0.f;
  for (int w = 0; w < nWaves; ++w) s += part[(size_t)w * DWN + j];
  if (j < 2400) {
    const int i = j / 50, o = j % 50;
    out[(i < 24) ? (i * 50 + o) : (1200 + (i - 24) * 50 + o)] = s;
  } else {
    const int t = j - 2400;
    const int i = t / 10, o = t % 10;
    out[(i < 24) ? (2400 + i * 10 + o) : (2640 + (i - 24) * 10 + o)] = s;
  }
}

__global__ void baesnn_zero(float* __restrict__ out) {
  const int j = blockIdx.x * blockDim.x + threadIdx.x;
  if (j < DWN) out[j] = 0.f;
}

extern "C" void kernel_launch(void* const* d_in, const int* in_sizes, int n_in,
                              void* d_out, int out_size, void* d_ws, size_t ws_size,
                              hipStream_t stream) {
  const float* x1 = (const float*)d_in[0];
  const float* x2 = (const float*)d_in[1];
  const float* W0 = (const float*)d_in[2];
  const float* W1 = (const float*)d_in[3];
  const float* W2 = (const float*)d_in[4];
  const float* W3 = (const float*)d_in[5];
  const float* W4 = (const float*)d_in[6];
  const float* W5 = (const float*)d_in[7];
  const float* W6 = (const float*)d_in[8];
  float* out = (float*)d_out;

  const int Bsz = in_sizes[0] / 24;            // [B,24] spike input
  const int nWaves = NBLOCKS * WPB;
  const size_t need = (size_t)nWaves * DWN * sizeof(float);

  if (ws_size >= need) {
    float* partial = (float*)d_ws;
    baesnn_main<<<NBLOCKS, TPB, 0, stream>>>(x1, x2, W0, W1, W2, W3, W4, W5, W6,
                                             out, partial, 0, Bsz, nWaves);
    baesnn_reduce<<<(DWN + 127) / 128, 128, 0, stream>>>(partial, out, nWaves);
  } else {
    baesnn_zero<<<(DWN + 127) / 128, 128, 0, stream>>>(out);
    baesnn_main<<<NBLOCKS, TPB, 0, stream>>>(x1, x2, W0, W1, W2, W3, W4, W5, W6,
                                             out, nullptr, 1, Bsz, nWaves);
  }
}
